// ParallelSelectiveMLP_80994493268150
// MI455X (gfx1250) — compile-verified, matrix operand order checked by reading, not run on hardware
//
#include <hip/hip_runtime.h>

// ---------------------------------------------------------------------------
// ParallelSelectiveMLP for MI455X (gfx1250, wave32, WMMA bf16 16x16x32)
//
//   h   = relu(x @ gather(fc1_w, idx)^T + gather(bias1, idx))   [T x K]
//   out = h @ gather(fc2_w_t, idx) + bias2                      [T x O]
//
// Strategy: convert operands to bf16 once (gather fused), transpose W2 so
// both GEMMs consume B as [N][K] row-major (K contiguous), then run a tiled
// v_wmma_f32_16x16x32_bf16 GEMM with f32 accumulation and fused epilogues.
//
// GEMM tiling: block tile 256x128 (8 waves, 4(M) x 2(N)); wave tile 64x64 =
// 4x4 WMMA 16x16 tiles -> 16 WMMAs per 8 LDS fragment loads per K-step.
// ---------------------------------------------------------------------------

typedef __bf16 bf16_t;
typedef __attribute__((ext_vector_type(16))) __bf16 v16bf;
typedef __attribute__((ext_vector_type(8)))  __bf16 v8bf;
typedef __attribute__((ext_vector_type(4)))  __bf16 v4bf;
typedef __attribute__((ext_vector_type(8)))  float  v8f;

#define TOKENS  4096
#define DMODEL  4096
#define DFF     16384
#define KSEL    4096

#define BM 256
#define BN 128
#define BK 32
#define LDK (BK + 8)   // pad LDS rows (80B stride) to spread banks

union FragBF {
    v16bf v;
    v8bf  h[2];
};

// ---------------------------------------------------------------------------
// Prep kernels
// ---------------------------------------------------------------------------

// fp32 -> bf16 bulk convert (vectorized float4 in / v4bf out)
__global__ __launch_bounds__(256)
void psm_cvt_f32_bf16(const float* __restrict__ src, bf16_t* __restrict__ dst) {
    int i = blockIdx.x * 256 + threadIdx.x;           // one float4 per thread
    float4 f = ((const float4*)src)[i];
    v4bf o;
    o[0] = (bf16_t)f.x; o[1] = (bf16_t)f.y; o[2] = (bf16_t)f.z; o[3] = (bf16_t)f.w;
    ((v4bf*)dst)[i] = o;
}

// gather rows of fc1_w by idx, convert to bf16; also gather bias1
__global__ __launch_bounds__(256)
void psm_gather_w1(const float* __restrict__ fc1, const float* __restrict__ bias1,
                   const int* __restrict__ idx, bf16_t* __restrict__ w1g,
                   float* __restrict__ b1g) {
    const int k = blockIdx.y;
    const int r = idx[k];
    const int c = (blockIdx.x * 256 + threadIdx.x) * 4;
    float4 f = *(const float4*)(fc1 + (size_t)r * DMODEL + c);
    v4bf o;
    o[0] = (bf16_t)f.x; o[1] = (bf16_t)f.y; o[2] = (bf16_t)f.z; o[3] = (bf16_t)f.w;
    *(v4bf*)(w1g + (size_t)k * DMODEL + c) = o;
    if (blockIdx.x == 0 && threadIdx.x == 0) b1g[k] = bias1[r];
}

// gather rows of fc2_w_t by idx, convert to bf16, transpose into w2gT[o][k]
// so the second GEMM's B operand has its reduction dim (k) contiguous.
__global__ __launch_bounds__(256)
void psm_gather_tr_w2(const float* __restrict__ fc2t, const int* __restrict__ idx,
                      bf16_t* __restrict__ w2gT) {
    __shared__ bf16_t tile[32][33];
    const int k0 = blockIdx.x * 32;
    const int o0 = blockIdx.y * 32;
    const int tx = threadIdx.x;          // 0..31
    const int ty = threadIdx.y;          // 0..7
#pragma unroll
    for (int j = 0; j < 4; ++j) {
        int k = k0 + ty + j * 8;
        int r = idx[k];
        tile[ty + j * 8][tx] = (bf16_t)fc2t[(size_t)r * DMODEL + o0 + tx];
    }
    __syncthreads();
#pragma unroll
    for (int j = 0; j < 4; ++j) {
        int o = o0 + ty + j * 8;
        w2gT[(size_t)o * KSEL + k0 + tx] = tile[tx][ty + j * 8];
    }
}

// ---------------------------------------------------------------------------
// WMMA bf16 GEMM:  D[M][N] = epilogue( A[M][K] * B[N][K]^T + bias[N] )
// A, B row-major with K contiguous. OUT_T in {bf16_t, float}.
// ---------------------------------------------------------------------------
template <typename OUT_T, bool RELU>
__global__ __launch_bounds__(256)
void psm_wmma_gemm(const bf16_t* __restrict__ A, const bf16_t* __restrict__ B,
                   const float* __restrict__ bias, OUT_T* __restrict__ D,
                   int N, int K) {
    __shared__ bf16_t As[BM][LDK];   // 256 x 40 bf16 = 20 KB
    __shared__ bf16_t Bs[BN][LDK];   // 128 x 40 bf16 = 10 KB

    const int tid  = threadIdx.x;
    const int lane = tid & 31;
    const int wave = tid >> 5;
    const int wm   = (wave >> 1) * 64;  // 4 waves along M
    const int wn   = (wave & 1) * 64;   // 2 waves along N
    const int m0   = blockIdx.y * BM;
    const int n0   = blockIdx.x * BN;

    // 16-bit A/B fragment layout (ISA 7.12.2): lanes 0-15 hold K 0-7 & 16-23,
    // lanes 16-31 hold K 8-15 & 24-31; row/col index = lane & 15.
    const int frow = lane & 15;
    const int kb   = (lane < 16) ? 0 : 8;

    v8f acc[4][4] = {};

    for (int kt = 0; kt < K; kt += BK) {
        // Stage A: 256x32 bf16 (1024 16B-chunks, 4 per thread)
#pragma unroll
        for (int j = 0; j < 4; ++j) {
            int c  = tid + j * 256;
            int r  = c >> 2;
            int cc = (c & 3) * 8;
            *(float4*)&As[r][cc] = *(const float4*)(A + (size_t)(m0 + r) * K + kt + cc);
        }
        // Stage B: 128x32 bf16 (512 16B-chunks, 2 per thread)
#pragma unroll
        for (int j = 0; j < 2; ++j) {
            int c  = tid + j * 256;
            int r  = c >> 2;
            int cc = (c & 3) * 8;
            *(float4*)&Bs[r][cc] = *(const float4*)(B + (size_t)(n0 + r) * K + kt + cc);
        }
        __syncthreads();

        FragBF a[4], b[4];
#pragma unroll
        for (int i = 0; i < 4; ++i) {
            a[i].h[0] = *(const v8bf*)&As[wm + i * 16 + frow][kb];
            a[i].h[1] = *(const v8bf*)&As[wm + i * 16 + frow][kb + 16];
        }
#pragma unroll
        for (int j = 0; j < 4; ++j) {
            b[j].h[0] = *(const v8bf*)&Bs[wn + j * 16 + frow][kb];
            b[j].h[1] = *(const v8bf*)&Bs[wn + j * 16 + frow][kb + 16];
        }

#pragma unroll
        for (int i = 0; i < 4; ++i)
#pragma unroll
            for (int j = 0; j < 4; ++j)
                acc[i][j] = __builtin_amdgcn_wmma_f32_16x16x32_bf16(
                    false, a[i].v, false, b[j].v, (short)0, acc[i][j], false, false);

        __syncthreads();
    }

    // Epilogue. C/D layout: lane<16 -> M = vgpr (0..7), N = lane;
    //                       lane>=16 -> M = 8+vgpr,    N = lane-16.
    const int nloc = lane & 15;
    const int mb   = (lane < 16) ? 0 : 8;
#pragma unroll
    for (int j = 0; j < 4; ++j) {
        const int   gn = n0 + wn + j * 16 + nloc;
        const float bv = bias[gn];
#pragma unroll
        for (int i = 0; i < 4; ++i) {
            const int gm = m0 + wm + i * 16 + mb;
#pragma unroll
            for (int e = 0; e < 8; ++e) {
                float v = acc[i][j][e] + bv;
                if (RELU) v = fmaxf(v, 0.0f);
                D[(size_t)(gm + e) * N + gn] = (OUT_T)v;
            }
        }
    }
}

// ---------------------------------------------------------------------------
// Launch
// ---------------------------------------------------------------------------
extern "C" void kernel_launch(void* const* d_in, const int* in_sizes, int n_in,
                              void* d_out, int out_size, void* d_ws, size_t ws_size,
                              hipStream_t stream) {
    (void)in_sizes; (void)n_in; (void)out_size; (void)ws_size;

    const float* x     = (const float*)d_in[0];   // [T, D]
    const float* fc1_w = (const float*)d_in[1];   // [DFF, D]
    const float* bias1 = (const float*)d_in[2];   // [DFF]
    const float* fc2_t = (const float*)d_in[3];   // [DFF, D]
    const float* bias2 = (const float*)d_in[4];   // [D]
    const int*   idx   = (const int*)d_in[5];     // [KSEL]
    float*       out   = (float*)d_out;           // [T, D]

    // Workspace layout (bf16 intermediates + gathered bias), ~134 MB total.
    char*   ws   = (char*)d_ws;
    bf16_t* xb   = (bf16_t*)ws;  ws += (size_t)TOKENS * DMODEL * sizeof(bf16_t);
    bf16_t* w1g  = (bf16_t*)ws;  ws += (size_t)KSEL   * DMODEL * sizeof(bf16_t);
    bf16_t* w2gT = (bf16_t*)ws;  ws += (size_t)DMODEL * KSEL   * sizeof(bf16_t);
    bf16_t* h    = (bf16_t*)ws;  ws += (size_t)TOKENS * KSEL   * sizeof(bf16_t);
    float*  b1g  = (float*)ws;

    // 1) x -> bf16
    psm_cvt_f32_bf16<<<(TOKENS * DMODEL) / (256 * 4), 256, 0, stream>>>(x, xb);

    // 2) gather W1 rows (+bias1) -> bf16
    psm_gather_w1<<<dim3(DMODEL / (256 * 4), KSEL), 256, 0, stream>>>(
        fc1_w, bias1, idx, w1g, b1g);

    // 3) gather + transpose W2 -> bf16 [O][K]
    psm_gather_tr_w2<<<dim3(KSEL / 32, DMODEL / 32), dim3(32, 8), 0, stream>>>(
        fc2_t, idx, w2gT);

    // 4) h = relu(x @ W1g^T + b1g)   (bf16 out)
    psm_wmma_gemm<bf16_t, true><<<dim3(KSEL / BN, TOKENS / BM), 256, 0, stream>>>(
        xb, w1g, b1g, h, KSEL, DMODEL);

    // 5) out = h @ W2g + bias2       (f32 out)
    psm_wmma_gemm<float, false><<<dim3(DMODEL / BN, TOKENS / BM), 256, 0, stream>>>(
        h, w2gT, bias2, out, DMODEL, KSEL);
}
